// RelativeAttention_18124761989462
// MI455X (gfx1250) — compile-verified, matrix operand order checked by reading, not run on hardware
//
#include <hip/hip_runtime.h>

typedef __attribute__((ext_vector_type(16))) _Float16 v16h;
typedef __attribute__((ext_vector_type(8)))  _Float16 v8h;
typedef __attribute__((ext_vector_type(8)))  float    v8f;
typedef __attribute__((ext_vector_type(4)))  int      v4i;

constexpr int BATCH = 4;
constexpr int SEQ   = 1024;   // query tokens == key tokens
constexpr int EMB_D = 1024;
constexpr int NHEAD = 8;
constexpr int HDIM  = 128;

__device__ __forceinline__ v8f wmma_f16(v16h a, v16h b, v8f c) {
  // D = A(16x32 f16) * B(32x16 f16) + C(16x16 f32)
  return __builtin_amdgcn_wmma_f32_16x16x32_f16(false, a, false, b, (short)0, c,
                                                false, false);
}

// ---- gfx1250 async global->LDS copy (ASYNCcnt path), 16B per lane ----------
__device__ __forceinline__ void cp_async_b128(const void* g, void* l) {
#if __has_builtin(__builtin_amdgcn_global_load_async_to_lds_b128)
  __builtin_amdgcn_global_load_async_to_lds_b128(
      (v4i*)g, (__attribute__((address_space(3))) v4i*)l, 0, 0);
#else
  *(v4i*)l = *(const v4i*)g;   // sync fallback through VGPRs
#endif
}

__device__ __forceinline__ void wait_async0() {
#if __has_builtin(__builtin_amdgcn_s_wait_asynccnt)
  __builtin_amdgcn_s_wait_asynccnt(0);
#else
  asm volatile("s_wait_asynccnt 0" ::: "memory");
#endif
}

// Fragment loader for row-major f16 in global memory (A-frags: rows=M;
// B-frags when matrix supplied transposed: rows=N, K contiguous).
__device__ __forceinline__ v16h ld_frag_f16(const _Float16* __restrict__ p,
                                            int ld, int r0, int k0, int lane) {
  const int row = r0 + (lane & 15);
  const int kb  = k0 + ((lane >> 4) << 3);
  const _Float16* q = p + (size_t)row * ld + kb;
  v8h lo = *(const v8h*)(q);
  v8h hi = *(const v8h*)(q + 16);
  v16h r;
#pragma unroll
  for (int e = 0; e < 8; ++e) { r[e] = lo[e]; r[e + 8] = hi[e]; }
  return r;
}

// ---------------------------------------------------------------------------
// Weight cast + transpose: Wt[n][k] = (f16) W[k][n]   (K x N row-major input)
// ---------------------------------------------------------------------------
__global__ __launch_bounds__(256)
void wcast_t(const float* __restrict__ W, _Float16* __restrict__ Wt,
             int K, int N) {
  __shared__ float tile[32][33];
  const int n0 = blockIdx.x * 32, k0 = blockIdx.y * 32;
  const int tx = threadIdx.x, ty = threadIdx.y;  // 32 x 8
#pragma unroll
  for (int yy = 0; yy < 32; yy += 8)
    tile[ty + yy][tx] = W[(size_t)(k0 + ty + yy) * N + n0 + tx];
  __syncthreads();
#pragma unroll
  for (int yy = 0; yy < 32; yy += 8)
    Wt[(size_t)(n0 + ty + yy) * K + k0 + tx] = (_Float16)tile[tx][ty + yy];
}

// ---------------------------------------------------------------------------
// GEMM: C(MxN) = A(MxK) * Bt(NxK)^T  [+ bias], async-LDS staged.
//   Block tile 128x256, K-step 32; 8 waves, each computes 64x64 (4x4 frags).
//   A/B tiles land in LDS via GLOBAL_LOAD_ASYNC_TO_LDS_B128; padded strides
//   spread the 16-lane fragment reads over 16 distinct LDS banks.
//   OMODE 0: C = f16 row-major
//   OMODE 1: C = f16 per-head transposed -> Vt[((b*NHEAD+h)*HDIM+d)*SEQ + j]
//   OMODE 2: C = f32 row-major + bias[n]
// ---------------------------------------------------------------------------
constexpr int BM = 128, BN = 256, BK = 32;

template <bool A_IS_F32, int OMODE>
__global__ __launch_bounds__(256)
void gemm_wmma(const void* __restrict__ Ap, const _Float16* __restrict__ Bt,
               const float* __restrict__ bias, void* __restrict__ Cp,
               int M, int N, int K) {
  constexpr int ASTR   = A_IS_F32 ? (BK + 4) : (BK + 8);      // elements
  constexpr int ABYTES = A_IS_F32 ? BM * ASTR * 4 : BM * ASTR * 2;
  constexpr int BSTR   = BK + 8;                              // halves
  __shared__ __align__(16) char smem[ABYTES + BN * BSTR * 2];
  float*    sAf = (float*)smem;
  _Float16* sAh = (_Float16*)smem;
  _Float16* sB  = (_Float16*)(smem + ABYTES);

  const int t      = threadIdx.x;
  const int lane   = t & 31;
  const int wave   = t >> 5;
  const int lanelo = lane & 15;
  const int lhi    = lane >> 4;
  const int gm0 = blockIdx.x * BM;
  const int gn0 = blockIdx.y * BN;
  const int wm  = (wave & 1) * 64;    // wave tile inside block
  const int wn  = (wave >> 1) * 64;

  const int arow  = t & (BM - 1);     // cooperative-copy row for A
  const int ahalf = t >> 7;           // which 16-element half of the row

  v8f acc[4][4];
#pragma unroll
  for (int x = 0; x < 4; ++x)
#pragma unroll
    for (int y = 0; y < 4; ++y)
#pragma unroll
      for (int e = 0; e < 8; ++e) acc[x][y][e] = 0.0f;

  for (int k0 = 0; k0 < K; k0 += BK) {
    // ---- stage A tile (128 x 32) and B tile (256 x 32) into LDS ----
    if (A_IS_F32) {
      const float* ag = (const float*)Ap + (size_t)(gm0 + arow) * K + k0 + ahalf * 16;
      float* al = sAf + arow * ASTR + ahalf * 16;
#pragma unroll
      for (int s = 0; s < 4; ++s) cp_async_b128(ag + s * 4, al + s * 4);
    } else {
      const _Float16* ag =
          (const _Float16*)Ap + (size_t)(gm0 + arow) * K + k0 + ahalf * 16;
      _Float16* al = sAh + arow * ASTR + ahalf * 16;
#pragma unroll
      for (int s = 0; s < 2; ++s) cp_async_b128(ag + s * 8, al + s * 8);
    }
    {
      const _Float16* bg = Bt + (size_t)(gn0 + t) * K + k0;
      _Float16* bl = sB + t * BSTR;
#pragma unroll
      for (int s = 0; s < 4; ++s) cp_async_b128(bg + s * 8, bl + s * 8);
    }
    wait_async0();
    __syncthreads();

    // ---- fragments from LDS ----
    v16h af[4], bf[4];
    const int kb = lhi << 3;
#pragma unroll
    for (int x = 0; x < 4; ++x) {
      const int row = wm + 16 * x + lanelo;
      if (A_IS_F32) {
        const float* q = sAf + row * ASTR + kb;
        v16h r;
#pragma unroll
        for (int e = 0; e < 8; ++e) {
          r[e]     = (_Float16)q[e];
          r[e + 8] = (_Float16)q[e + 16];
        }
        af[x] = r;
      } else {
        const _Float16* q = sAh + row * ASTR + kb;
        v8h lo = *(const v8h*)q;
        v8h hi = *(const v8h*)(q + 16);
        v16h r;
#pragma unroll
        for (int e = 0; e < 8; ++e) { r[e] = lo[e]; r[e + 8] = hi[e]; }
        af[x] = r;
      }
    }
#pragma unroll
    for (int y = 0; y < 4; ++y) {
      const _Float16* q = sB + (wn + 16 * y + lanelo) * BSTR + kb;
      v8h lo = *(const v8h*)q;
      v8h hi = *(const v8h*)(q + 16);
      v16h r;
#pragma unroll
      for (int e = 0; e < 8; ++e) { r[e] = lo[e]; r[e + 8] = hi[e]; }
      bf[y] = r;
    }
#pragma unroll
    for (int x = 0; x < 4; ++x)
#pragma unroll
      for (int y = 0; y < 4; ++y)
        acc[x][y] = wmma_f16(af[x], bf[y], acc[x][y]);
    __syncthreads();
  }

  // ---- epilogue ----
  const int m0 = gm0 + wm;
  const int n0 = gn0 + wn;
  if (OMODE == 0) {
    _Float16* C = (_Float16*)Cp;
#pragma unroll
    for (int x = 0; x < 4; ++x)
#pragma unroll
      for (int y = 0; y < 4; ++y) {
        const int n = n0 + 16 * y + lanelo;
#pragma unroll
        for (int r = 0; r < 8; ++r) {
          const int m = m0 + 16 * x + 8 * lhi + r;
          C[(size_t)m * N + n] = (_Float16)acc[x][y][r];
        }
      }
  } else if (OMODE == 1) {
    _Float16* C = (_Float16*)Cp;
#pragma unroll
    for (int x = 0; x < 4; ++x) {
      const int mb = m0 + 16 * x;          // fragment row base (j direction)
      const int b  = mb / SEQ;             // tiles never cross batch boundary
      const int j  = (mb % SEQ) + 8 * lhi; // 8 consecutive j per lane
#pragma unroll
      for (int y = 0; y < 4; ++y) {
        const int n = n0 + 16 * y + lanelo;
        const int h = n >> 7, d = n & (HDIM - 1);
        v8h st;
#pragma unroll
        for (int r = 0; r < 8; ++r) st[r] = (_Float16)acc[x][y][r];
        *(v8h*)(C + ((size_t)(b * NHEAD + h) * HDIM + d) * SEQ + j) = st;
      }
    }
  } else {
    float* C = (float*)Cp;
#pragma unroll
    for (int y = 0; y < 4; ++y) {
      const int n   = n0 + 16 * y + lanelo;
      const float bv = bias[n];
#pragma unroll
      for (int x = 0; x < 4; ++x)
#pragma unroll
        for (int r = 0; r < 8; ++r) {
          const int m = m0 + 16 * x + 8 * lhi + r;
          C[(size_t)m * N + n] = acc[x][y][r] + bv;
        }
    }
  }
}

// ---------------------------------------------------------------------------
// Flash attention, transposed-S formulation (unchanged math, + prefetch).
//   S^T = K * Q^T -> lane = i (softmax row), element = j.
//   One shfl_xor(16) merges the j-halves; P^T fragment pair IS the
//   B-fragment for O^T += V^T * P^T (no LDS round trip).
// ---------------------------------------------------------------------------
__global__ __launch_bounds__(256)
void attn_flash(const _Float16* __restrict__ Qh,   // [B*SEQ][EMB]
                const _Float16* __restrict__ Kh,   // [B*SEQ][EMB]
                const _Float16* __restrict__ Vt,   // [B][H][HDIM][SEQ]
                const int* __restrict__ mask,      // [B][SEQ][SEQ]
                _Float16* __restrict__ AV) {       // [B*SEQ][EMB]
  const int lane   = threadIdx.x & 31;
  const int wave   = threadIdx.x >> 5;   // 8 waves/block
  const int lanelo = lane & 15;
  const int lhi    = lane >> 4;
  const int b  = blockIdx.z;
  const int h  = blockIdx.y;
  const int i0 = blockIdx.x * 128 + wave * 16;
  const int i  = i0 + lanelo;            // this lane's query row

  const _Float16* qbase = Qh + (size_t)(b * SEQ) * EMB_D + h * HDIM;
  const _Float16* kbase = Kh + (size_t)(b * SEQ) * EMB_D + h * HDIM;
  const _Float16* vbase = Vt + ((size_t)(b * NHEAD + h) * HDIM) * SEQ;
  const int* mbase = mask + ((size_t)b * SEQ + i) * SEQ;

  // Q B-fragments over the 128-wide head dim (4 x K=32), reused for all j.
  v16h qfrag[4];
#pragma unroll
  for (int kd = 0; kd < 4; ++kd)
    qfrag[kd] = ld_frag_f16(qbase, EMB_D, i0, kd * 32, lane);

  float m_run = -3.0e38f, l_run = 0.0f;
  v8f o[8];
#pragma unroll
  for (int f = 0; f < 8; ++f)
#pragma unroll
    for (int e = 0; e < 8; ++e) o[f][e] = 0.0f;

  for (int j0 = 0; j0 < SEQ; j0 += 32) {
    // hint the next j-tile of K and V into cache (global_prefetch_b8)
    if (j0 + 32 < SEQ) {
      __builtin_prefetch(kbase + (size_t)(j0 + 32 + lanelo) * EMB_D, 0, 1);
      __builtin_prefetch(kbase + (size_t)(j0 + 48 + lanelo) * EMB_D, 0, 1);
      __builtin_prefetch(vbase + (size_t)(lane << 2) * SEQ + j0 + 32, 0, 1);
    }
    // --- S^T for two 16-j tiles, accumulated over d ---
    v8f s0, s1;
#pragma unroll
    for (int e = 0; e < 8; ++e) { s0[e] = 0.0f; s1[e] = 0.0f; }
#pragma unroll
    for (int kd = 0; kd < 4; ++kd) {
      v16h ka0 = ld_frag_f16(kbase, EMB_D, j0,      kd * 32, lane);
      v16h ka1 = ld_frag_f16(kbase, EMB_D, j0 + 16, kd * 32, lane);
      s0 = wmma_f16(ka0, qfrag[kd], s0);
      s1 = wmma_f16(ka1, qfrag[kd], s1);
    }
    // --- mask + online softmax (reference applies no 1/sqrt(d) scale) ---
    const int* mp0 = mbase + j0 + 8 * lhi;       // 8 contiguous j's
    const int* mp1 = mp0 + 16;
    float sv0[8], sv1[8], tmax = -3.0e38f;
#pragma unroll
    for (int r = 0; r < 8; ++r) {
      const float x0 = (mp0[r] == 0) ? -1.0e9f : s0[r];
      const float x1 = (mp1[r] == 0) ? -1.0e9f : s1[r];
      sv0[r] = x0; sv1[r] = x1;
      tmax = fmaxf(tmax, fmaxf(x0, x1));
    }
    tmax = fmaxf(tmax, __shfl_xor(tmax, 16, 32));  // merge the two j-halves
    const float m_new = fmaxf(m_run, tmax);
    const float scale = __expf(m_run - m_new);
    float lsum = 0.0f;
    v16h pfrag;                                   // = B-fragment of P^T
#pragma unroll
    for (int r = 0; r < 8; ++r) {
      const float p0 = __expf(sv0[r] - m_new);
      const float p1 = __expf(sv1[r] - m_new);
      lsum += p0 + p1;
      pfrag[r]     = (_Float16)p0;
      pfrag[r + 8] = (_Float16)p1;
    }
    lsum += __shfl_xor(lsum, 16, 32);
    l_run = l_run * scale + lsum;
    m_run = m_new;
    // --- O^T += V^T * P^T over all 8 d-fragments ---
#pragma unroll
    for (int f = 0; f < 8; ++f) {
      v8f oo = o[f];
#pragma unroll
      for (int r = 0; r < 8; ++r) oo[r] *= scale;
      v16h va = ld_frag_f16(vbase, SEQ, f * 16, j0, lane);  // rows = d, K = j
      o[f] = wmma_f16(va, pfrag, oo);
    }
  }

  // --- epilogue: O / l, store f16; lane's 8 elements are contiguous in d ---
  const float inv = 1.0f / l_run;
  _Float16* obase = AV + ((size_t)(b * SEQ) + i) * EMB_D + h * HDIM;
#pragma unroll
  for (int f = 0; f < 8; ++f) {
    v8h st;
#pragma unroll
    for (int r = 0; r < 8; ++r) st[r] = (_Float16)(o[f][r] * inv);
    *(v8h*)(obase + f * 16 + 8 * lhi) = st;
  }
}

// ---------------------------------------------------------------------------
extern "C" void kernel_launch(void* const* d_in, const int* in_sizes, int n_in,
                              void* d_out, int out_size, void* d_ws,
                              size_t ws_size, hipStream_t stream) {
  const float* query = (const float*)d_in[0];
  const float* keyt  = (const float*)d_in[1];
  const int*   maskp = (const int*)d_in[2];
  const float* Wq = (const float*)d_in[3];
  const float* Wk = (const float*)d_in[4];
  const float* Wv = (const float*)d_in[5];
  const float* Wu = (const float*)d_in[6];
  const float* bu = (const float*)d_in[7];

  const size_t ACT = (size_t)BATCH * SEQ * EMB_D;  // 4Mi elements
  const size_t WEL = (size_t)EMB_D * EMB_D;        // 1Mi elements
  _Float16* ws  = (_Float16*)d_ws;                 // 40 MB total
  _Float16* Qh  = ws;
  _Float16* Kh  = Qh + ACT;
  _Float16* Vt  = Kh + ACT;
  _Float16* AV  = Vt + ACT;
  _Float16* Wqt = AV + ACT;
  _Float16* Wkt = Wqt + WEL;
  _Float16* Wvt = Wkt + WEL;
  _Float16* Wut = Wvt + WEL;

  dim3 tb(32, 8), tg(EMB_D / 32, EMB_D / 32);
  wcast_t<<<tg, tb, 0, stream>>>(Wq, Wqt, EMB_D, EMB_D);
  wcast_t<<<tg, tb, 0, stream>>>(Wk, Wkt, EMB_D, EMB_D);
  wcast_t<<<tg, tb, 0, stream>>>(Wv, Wvt, EMB_D, EMB_D);
  wcast_t<<<tg, tb, 0, stream>>>(Wu, Wut, EMB_D, EMB_D);

  const int M = BATCH * SEQ;
  dim3 gb(256), gg(M / BM, EMB_D / BN);
  gemm_wmma<true, 0><<<gg, gb, 0, stream>>>(query, Wqt, nullptr, Qh, M, EMB_D, EMB_D);
  gemm_wmma<true, 0><<<gg, gb, 0, stream>>>(keyt,  Wkt, nullptr, Kh, M, EMB_D, EMB_D);
  gemm_wmma<true, 1><<<gg, gb, 0, stream>>>(keyt,  Wvt, nullptr, Vt, M, EMB_D, EMB_D);

  dim3 ag(SEQ / 128, NHEAD, BATCH);
  attn_flash<<<ag, dim3(256), 0, stream>>>(Qh, Kh, Vt, maskp, AV);

  gemm_wmma<false, 2><<<gg, gb, 0, stream>>>(AV, Wut, bu, d_out, M, EMB_D, EMB_D);
}
